// Neural3DHMM_64707977281582
// MI455X (gfx1250) — matrix operand order repaired and checked by reading, not compiled, not run on hardware
//
#include <hip/hip_runtime.h>
#include <math.h>

typedef __attribute__((ext_vector_type(16))) __bf16    v16bf;
typedef __attribute__((ext_vector_type(8)))  __bf16    v8bf;
typedef __attribute__((ext_vector_type(16))) _Float16  v16h;
typedef __attribute__((ext_vector_type(8)))  _Float16  v8h;
typedef __attribute__((ext_vector_type(8)))  float     v8f;

#define S_TOT 2048
#define XYDIM 16
#define VTOK 10000
#define EMB 256
#define TOKD 100
#define TOKP 128   // emit hidden dim padded to 128 (multiple of 32)
#define BB 8
#define TT 16
#define LL 12
#define NTILES_V (VTOK / 16)  // 625 N-tiles in the emission GEMM

// ---------------- branch-free WMMA fragment loaders (pre-converted 16-bit data) ----------------
// A fragment 16x32 (MxK): lane<16 -> row=lane,   K in {kb..kb+7, kb+16..kb+23}, kb=kBase
//                         lane>=16 -> row=lane-16, kb=kBase+8
__device__ inline v16bf load_a_bf(const __bf16* A, int lda, int mBase, int kBase) {
  const int lane = threadIdx.x & 31;
  const __bf16* p = A + (size_t)(mBase + (lane & 15)) * lda + kBase + ((lane >> 4) << 3);
  v8bf lo = *(const v8bf*)p;
  v8bf hi = *(const v8bf*)(p + 16);
  return __builtin_shufflevector(lo, hi, 0, 1, 2, 3, 4, 5, 6, 7, 8, 9, 10, 11, 12, 13, 14, 15);
}
// B fragment 32x16 (KxN) from W[N,K] row-major: lane<16 -> col=lane,   K in {kb..kb+7, kb+16..kb+23}, kb=kBase
//                                               lane>=16 -> col=lane-16, kb=kBase+8
__device__ inline v16bf load_b_bf(const __bf16* W, int ldw, int nBase, int kBase) {
  const int lane = threadIdx.x & 31;
  const __bf16* p = W + (size_t)(nBase + (lane & 15)) * ldw + kBase + ((lane >> 4) << 3);
  v8bf lo = *(const v8bf*)p;
  v8bf hi = *(const v8bf*)(p + 16);
  return __builtin_shufflevector(lo, hi, 0, 1, 2, 3, 4, 5, 6, 7, 8, 9, 10, 11, 12, 13, 14, 15);
}
__device__ inline v16h load_a_h(const _Float16* A, int lda, int mBase, int kBase) {
  const int lane = threadIdx.x & 31;
  const _Float16* p = A + (size_t)(mBase + (lane & 15)) * lda + kBase + ((lane >> 4) << 3);
  v8h lo = *(const v8h*)p;
  v8h hi = *(const v8h*)(p + 16);
  return __builtin_shufflevector(lo, hi, 0, 1, 2, 3, 4, 5, 6, 7, 8, 9, 10, 11, 12, 13, 14, 15);
}
__device__ inline v16h load_b_h(const _Float16* W, int ldw, int nBase, int kBase) {
  const int lane = threadIdx.x & 31;
  const _Float16* p = W + (size_t)(nBase + (lane & 15)) * ldw + kBase + ((lane >> 4) << 3);
  v8h lo = *(const v8h*)p;
  v8h hi = *(const v8h*)(p + 16);
  return __builtin_shufflevector(lo, hi, 0, 1, 2, 3, 4, 5, 6, 7, 8, 9, 10, 11, 12, 13, 14, 15);
}

// ---------------- convert / zero-pad helpers ----------------
__global__ __launch_bounds__(256) void cvt_pad_bf16_k(const float* __restrict__ src, int sR, int sC,
                                                      __bf16* __restrict__ dst, int dC, int total) {
  const int i = blockIdx.x * 256 + threadIdx.x;
  if (i >= total) return;
  const int r = i / dC, c = i - r * dC;
  const float v = (r < sR && c < sC) ? src[r * sC + c] : 0.f;
  dst[i] = (__bf16)v;
}
__global__ __launch_bounds__(256) void cvt_pad_f16_k(const float* __restrict__ src, int sR, int sC,
                                                     _Float16* __restrict__ dst, int dC, int total) {
  const int i = blockIdx.x * 256 + threadIdx.x;
  if (i >= total) return;
  const int r = i / dC, c = i - r * dC;
  const float v = (r < sR && c < sC) ? src[r * sC + c] : 0.f;
  dst[i] = (_Float16)v;
}
__global__ __launch_bounds__(256) void pad_vec_k(const float* __restrict__ src, int n,
                                                 float* __restrict__ dst, int N) {
  const int i = blockIdx.x * 256 + threadIdx.x;
  if (i >= N) return;
  dst[i] = (i < n) ? src[i] : 0.f;
}

// ---------------- Residual block: H = LN(relu(relu(X@W1^T+b1)@W2^T+b2) + relu1) ----------------
// One block per 16-row band of states; 4 waves tile dout with bf16 WMMA; all dims compile-time.
template <int DIN, int DOUT, int DOUTPAD>
__global__ __launch_bounds__(128) void residual_kernel(
    const __bf16* __restrict__ X,   // [2048, DIN]
    const __bf16* __restrict__ W1,  // [DOUTPAD, DIN]   (zero-padded rows)
    const float*  __restrict__ B1,  // [DOUTPAD]        (zero-padded)
    const __bf16* __restrict__ W2,  // [DOUTPAD, DOUTPAD]
    const float*  __restrict__ B2,  // [DOUTPAD]
    const float*  __restrict__ G, const float* __restrict__ Bt,  // [DOUT]
    float* __restrict__ H)          // [2048, DOUTPAD]
{
  __shared__ __align__(32) float  sX1f[16 * DOUTPAD];
  __shared__ __align__(32) __bf16 sX1b[16 * DOUTPAD];
  __shared__ __align__(32) float  sH[16 * DOUTPAD];
  __shared__ float sMu[16], sRs[16];
  const int tid   = threadIdx.x;
  const int wave  = tid >> 5;
  const int lane  = tid & 31;
  const int mBase = blockIdx.x * 16;
  const int col   = lane & 15;
  const int rbase = (lane < 16) ? 0 : 8;
  constexpr int NT = DOUTPAD / 16;

  // X1 = relu(X @ W1^T + b1) -> LDS (fp32 for residual add, bf16 for next GEMM)
  for (int nt = wave; nt < NT; nt += 4) {
    v8f c = {};
#pragma unroll
    for (int kt = 0; kt < DIN; kt += 32) {
      v16bf a = load_a_bf(X, DIN, mBase, kt);
      v16bf b = load_b_bf(W1, DIN, nt * 16, kt);
      c = __builtin_amdgcn_wmma_f32_16x16x32_bf16(false, a, false, b, (short)0, c, false, false);
    }
    const int n = nt * 16 + col;
    const float bias = B1[n];
#pragma unroll
    for (int r = 0; r < 8; ++r) {
      const float v = fmaxf(c[r] + bias, 0.f);
      sX1f[(rbase + r) * DOUTPAD + n] = v;
      sX1b[(rbase + r) * DOUTPAD + n] = (__bf16)v;
    }
  }
  __syncthreads();

  // X2 = relu(X1 @ W2^T + b2); h = x2 + x1 -> LDS
  for (int nt = wave; nt < NT; nt += 4) {
    v8f c = {};
#pragma unroll
    for (int kt = 0; kt < DOUTPAD; kt += 32) {
      v16bf a = load_a_bf(sX1b, DOUTPAD, 0, kt);
      v16bf b = load_b_bf(W2, DOUTPAD, nt * 16, kt);
      c = __builtin_amdgcn_wmma_f32_16x16x32_bf16(false, a, false, b, (short)0, c, false, false);
    }
    const int n = nt * 16 + col;
    const float bias = B2[n];
#pragma unroll
    for (int r = 0; r < 8; ++r) {
      const float x1v = sX1f[(rbase + r) * DOUTPAD + n];
      const float x2v = fmaxf(c[r] + bias, 0.f);
      sH[(rbase + r) * DOUTPAD + n] = x2v + x1v;
    }
  }
  __syncthreads();

  if (tid < 16) {
    float mu = 0.f;
    for (int cc = 0; cc < DOUT; ++cc) mu += sH[tid * DOUTPAD + cc];
    mu /= (float)DOUT;
    float var = 0.f;
    for (int cc = 0; cc < DOUT; ++cc) { const float d = sH[tid * DOUTPAD + cc] - mu; var += d * d; }
    var /= (float)DOUT;
    sMu[tid] = mu;
    sRs[tid] = rsqrtf(var + 1e-5f);
  }
  __syncthreads();

  for (int idx = tid; idx < 16 * DOUTPAD; idx += 128) {
    const int row = idx / DOUTPAD, cc = idx - row * DOUTPAD;
    float v = 0.f;
    if (cc < DOUT) v = (sH[idx] - sMu[row]) * sRs[row] * G[cc] + Bt[cc];
    H[(size_t)(mBase + row) * DOUTPAD + cc] = v;
  }
}

// ---------------- priors = log_softmax(h0 @ wo^T + bo) over 2048 states ----------------
__global__ __launch_bounds__(1024) void priors_kernel(const float* __restrict__ h0,
                                                      const float* __restrict__ wo,
                                                      const float* __restrict__ bo,
                                                      float* __restrict__ priors) {
  __shared__ float sl[S_TOT];
  __shared__ float red[1024];
  const int tid = threadIdx.x;
  for (int j = tid; j < S_TOT; j += 1024) {
    float d = bo[0];
    for (int c = 0; c < EMB; ++c) d += h0[j * EMB + c] * wo[c];
    sl[j] = d;
  }
  __syncthreads();
  float m = -INFINITY;
  for (int j = tid; j < S_TOT; j += 1024) m = fmaxf(m, sl[j]);
  red[tid] = m; __syncthreads();
  for (int o = 512; o > 0; o >>= 1) { if (tid < o) red[tid] = fmaxf(red[tid], red[tid + o]); __syncthreads(); }
  const float M = red[0]; __syncthreads();
  float s = 0.f;
  for (int j = tid; j < S_TOT; j += 1024) s += __expf(sl[j] - M);
  red[tid] = s; __syncthreads();
  for (int o = 512; o > 0; o >>= 1) { if (tid < o) red[tid] += red[tid + o]; __syncthreads(); }
  const float lse = M + __logf(red[0]);
  __syncthreads();
  for (int j = tid; j < S_TOT; j += 1024) priors[j] = sl[j] - lse;
}

// ---------------- sparse transition rows: trans7[j,k] = log_softmax over 7 neighbors ----------------
__global__ __launch_bounds__(256) void trans7_kernel(const float* __restrict__ hIn,
                                                     const float* __restrict__ hOut,
                                                     float* __restrict__ trans7) {
  const int j = blockIdx.x * blockDim.x + threadIdx.x;
  if (j >= S_TOT) return;
  const int offs[7] = {0, 1, -1, XYDIM, -XYDIM, XYDIM * XYDIM, 2 * XYDIM * XYDIM};
  float vals[7];
  float mx = -INFINITY;
#pragma unroll
  for (int k = 0; k < 7; ++k) {
    const int i = j + offs[k];
    float v = -INFINITY;
    if (i >= 0 && i < S_TOT) {
      float d = 0.f;
      for (int c = 0; c < EMB; ++c) d += hIn[j * EMB + c] * hOut[i * EMB + c];
      v = d;
    }
    vals[k] = v;
    mx = fmaxf(mx, v);
  }
  float sum = 0.f;
#pragma unroll
  for (int k = 0; k < 7; ++k) sum += __expf(vals[k] - mx);
  const float lse = mx + __logf(sum);
#pragma unroll
  for (int k = 0; k < 7; ++k)
    trans7[j * 8 + k] = (vals[k] > -INFINITY) ? vals[k] - lse : -INFINITY;
  trans7[j * 8 + 7] = -INFINITY;
}

// ---------------- emission logits GEMM (f16 WMMA, K=128, A register-blocked over 8 N-tiles) ----------------
__global__ __launch_bounds__(256) void emit_logits_kernel(const _Float16* __restrict__ A,     // [2048,128]
                                                          const _Float16* __restrict__ Btok,  // [10000,128]
                                                          _Float16* __restrict__ logits) {    // [2048,10000]
  const int wave = threadIdx.x >> 5;
  const int lane = threadIdx.x & 31;
  const int mBase = blockIdx.y * 16;
  const int nt0 = (blockIdx.x * 8 + wave) * 8;  // 8 consecutive N-tiles per wave
  if (nt0 >= NTILES_V) return;

  // Load the wave's 16x128 A band once into registers (4 k-fragments = 32 VGPRs),
  // then sweep N-tiles so only B streams from L2.
  v16h a0 = load_a_h(A, TOKP, mBase, 0);
  v16h a1 = load_a_h(A, TOKP, mBase, 32);
  v16h a2 = load_a_h(A, TOKP, mBase, 64);
  v16h a3 = load_a_h(A, TOKP, mBase, 96);

  const int n = lane & 15;
  const int rbase = (lane < 16) ? 0 : 8;
#pragma unroll
  for (int ti = 0; ti < 8; ++ti) {
    const int nt = nt0 + ti;
    if (nt >= NTILES_V) break;
    const int nBase = nt * 16;
    v8f c = {};
    c = __builtin_amdgcn_wmma_f32_16x16x32_f16(false, a0, false, load_b_h(Btok, TOKP, nBase, 0),  (short)0, c, false, false);
    c = __builtin_amdgcn_wmma_f32_16x16x32_f16(false, a1, false, load_b_h(Btok, TOKP, nBase, 32), (short)0, c, false, false);
    c = __builtin_amdgcn_wmma_f32_16x16x32_f16(false, a2, false, load_b_h(Btok, TOKP, nBase, 64), (short)0, c, false, false);
    c = __builtin_amdgcn_wmma_f32_16x16x32_f16(false, a3, false, load_b_h(Btok, TOKP, nBase, 96), (short)0, c, false, false);
#pragma unroll
    for (int r = 0; r < 8; ++r)
      logits[(size_t)(mBase + rbase + r) * VTOK + nBase + n] = (_Float16)c[r];
  }
}

// ---------------- per-state softmax normalizer over 10000 tokens (vectorized fp16 reads) ----------------
__global__ __launch_bounds__(256) void logz_kernel(const _Float16* __restrict__ logits,
                                                   float* __restrict__ logZ) {
  __shared__ float red[256];
  const int s = blockIdx.x;
  const int tid = threadIdx.x;
  const v8h* row = (const v8h*)(logits + (size_t)s * VTOK);  // 10000 = 1250 * 8
  float m = -INFINITY;
  for (int v = tid; v < VTOK / 8; v += 256) {
    v8h x = row[v];
#pragma unroll
    for (int i = 0; i < 8; ++i) m = fmaxf(m, (float)x[i]);
  }
  red[tid] = m; __syncthreads();
  for (int o = 128; o > 0; o >>= 1) { if (tid < o) red[tid] = fmaxf(red[tid], red[tid + o]); __syncthreads(); }
  const float M = red[0]; __syncthreads();
  float sum = 0.f;
  for (int v = tid; v < VTOK / 8; v += 256) {
    v8h x = row[v];
#pragma unroll
    for (int i = 0; i < 8; ++i) sum += __expf((float)x[i] - M);
  }
  red[tid] = sum; __syncthreads();
  for (int o = 128; o > 0; o >>= 1) { if (tid < o) red[tid] += red[tid + o]; __syncthreads(); }
  if (tid == 0) logZ[s] = M + __logf(red[0]);
}

// ---------------- emis[b,t,s] = sum_l pooled-log-prob of story token (5x5 edge-clamped window) ----------------
__global__ __launch_bounds__(256) void emis_kernel(const _Float16* __restrict__ logits,
                                                   const float* __restrict__ logZ,
                                                   const int* __restrict__ stories,
                                                   float* __restrict__ emis) {
  const int idx = blockIdx.x * blockDim.x + threadIdx.x;
  if (idx >= BB * TT * S_TOT) return;
  const int s  = idx & (S_TOT - 1);
  const int bt = idx >> 11;
  const int t  = bt & (TT - 1);
  const int b  = bt >> 4;
  const int z = s >> 8, x = (s >> 4) & 15, y = s & 15;
  float acc = 0.f;
  for (int l = 0; l < LL; ++l) {
    const int tok = stories[(b * TT + t) * LL + l];
    if (tok >= VTOK) continue;  // pad token contributes exactly 0
    float vals[25];
    float mx = -INFINITY;
    int c = 0;
#pragma unroll
    for (int dx = -2; dx <= 2; ++dx) {
      int xc = x + dx; xc = xc < 0 ? 0 : (xc > 15 ? 15 : xc);
#pragma unroll
      for (int dy = -2; dy <= 2; ++dy) {
        int yc = y + dy; yc = yc < 0 ? 0 : (yc > 15 ? 15 : yc);
        const int sp = (z * 16 + xc) * 16 + yc;
        const float v = (float)logits[(size_t)sp * VTOK + tok] - logZ[sp];
        vals[c++] = v;
        mx = fmaxf(mx, v);
      }
    }
    float sum = 0.f;
#pragma unroll
    for (int i = 0; i < 25; ++i) sum += __expf(vals[i] - mx);
    acc += mx + __logf(sum * (1.0f / 25.0f));
  }
  emis[idx] = acc;
}

// ---------------- HMM forward scan: whole state [8,2048] lives in LDS of one WGP ----------------
__global__ __launch_bounds__(1024) void forward_kernel(const float* __restrict__ emis,
                                                       const float* __restrict__ priors,
                                                       const float* __restrict__ trans7,
                                                       float* __restrict__ out) {
  __shared__ float sc[BB * S_TOT];  // 64 KB, fits 320 KB WGP LDS
  const int tid = threadIdx.x;
  const int offs[7] = {0, 1, -1, XYDIM, -XYDIM, XYDIM * XYDIM, 2 * XYDIM * XYDIM};
  for (int idx = tid; idx < BB * S_TOT; idx += 1024) {
    const int b = idx >> 11, j = idx & (S_TOT - 1);
    const float v = emis[(b * TT + 0) * S_TOT + j] + priors[j];
    sc[idx] = v;
    out[(0 * BB + b) * S_TOT + j] = v;
  }
  __syncthreads();
  for (int t = 1; t < TT; ++t) {
    float nv[16];
    int c = 0;
    for (int idx = tid; idx < BB * S_TOT; idx += 1024, ++c) {
      const int b = idx >> 11, j = idx & (S_TOT - 1);
      float vals[7];
      float mx = -INFINITY;
#pragma unroll
      for (int k = 0; k < 7; ++k) {
        const int i = j + offs[k];
        float v = -INFINITY;
        if (i >= 0 && i < S_TOT) v = trans7[j * 8 + k] + sc[b * S_TOT + i];
        vals[k] = v;
        mx = fmaxf(mx, v);
      }
      float sum = 0.f;
#pragma unroll
      for (int k = 0; k < 7; ++k) sum += __expf(vals[k] - mx);
      nv[c] = emis[(b * TT + t) * S_TOT + j] + mx + __logf(sum);
    }
    __syncthreads();
    c = 0;
    for (int idx = tid; idx < BB * S_TOT; idx += 1024, ++c) {
      sc[idx] = nv[c];
      const int b = idx >> 11, j = idx & (S_TOT - 1);
      out[(t * BB + b) * S_TOT + j] = nv[c];
    }
    __syncthreads();
  }
}

extern "C" void kernel_launch(void* const* d_in, const int* in_sizes, int n_in,
                              void* d_out, int out_size, void* d_ws, size_t ws_size,
                              hipStream_t stream) {
  (void)in_sizes; (void)n_in; (void)out_size; (void)ws_size;
  const float* state_emb  = (const float*)d_in[0];
  const float* tok_emb    = (const float*)d_in[1];
  const float* start_w1   = (const float*)d_in[2];
  const float* start_b1   = (const float*)d_in[3];
  const float* start_w2   = (const float*)d_in[4];
  const float* start_b2   = (const float*)d_in[5];
  const float* start_g    = (const float*)d_in[6];
  const float* start_beta = (const float*)d_in[7];
  const float* start_wo   = (const float*)d_in[8];
  const float* start_bo   = (const float*)d_in[9];
  const float* tin_w1     = (const float*)d_in[10];
  const float* tin_b1     = (const float*)d_in[11];
  const float* tin_w2     = (const float*)d_in[12];
  const float* tin_b2     = (const float*)d_in[13];
  const float* tin_g      = (const float*)d_in[14];
  const float* tin_beta   = (const float*)d_in[15];
  const float* tout_w1    = (const float*)d_in[16];
  const float* tout_b1    = (const float*)d_in[17];
  const float* tout_w2    = (const float*)d_in[18];
  const float* tout_b2    = (const float*)d_in[19];
  const float* tout_g     = (const float*)d_in[20];
  const float* tout_beta  = (const float*)d_in[21];
  const float* emit_w1    = (const float*)d_in[22];
  const float* emit_b1    = (const float*)d_in[23];
  const float* emit_w2    = (const float*)d_in[24];
  const float* emit_b2    = (const float*)d_in[25];
  const float* emit_g     = (const float*)d_in[26];
  const float* emit_beta  = (const float*)d_in[27];
  const int*   stories    = (const int*)d_in[28];

  char* ws = (char*)d_ws;
  size_t off = 0;
  auto take = [&](size_t n) -> char* {
    char* p = ws + off;
    off += (n + 255) & ~(size_t)255;
    return p;
  };
  // fp32 activation buffers
  float*    h0      = (float*)take((size_t)S_TOT * EMB * 4);
  float*    hin     = (float*)take((size_t)S_TOT * EMB * 4);
  float*    hout    = (float*)take((size_t)S_TOT * EMB * 4);
  float*    hemit   = (float*)take((size_t)S_TOT * TOKP * 4);
  float*    priors  = (float*)take((size_t)S_TOT * 4);
  float*    trans7  = (float*)take((size_t)S_TOT * 8 * 4);
  float*    logZ    = (float*)take((size_t)S_TOT * 4);
  float*    emis    = (float*)take((size_t)BB * TT * S_TOT * 4);
  // pre-converted 16-bit GEMM operands
  __bf16*   Xbf     = (__bf16*)take((size_t)S_TOT * EMB * 2);
  __bf16*   sW1     = (__bf16*)take((size_t)EMB * EMB * 2);
  __bf16*   sW2     = (__bf16*)take((size_t)EMB * EMB * 2);
  __bf16*   iW1     = (__bf16*)take((size_t)EMB * EMB * 2);
  __bf16*   iW2     = (__bf16*)take((size_t)EMB * EMB * 2);
  __bf16*   oW1     = (__bf16*)take((size_t)EMB * EMB * 2);
  __bf16*   oW2     = (__bf16*)take((size_t)EMB * EMB * 2);
  __bf16*   eW1     = (__bf16*)take((size_t)TOKP * EMB * 2);
  __bf16*   eW2     = (__bf16*)take((size_t)TOKP * TOKP * 2);
  float*    eB1     = (float*)take((size_t)TOKP * 4);
  float*    eB2     = (float*)take((size_t)TOKP * 4);
  _Float16* tokh    = (_Float16*)take((size_t)VTOK * TOKP * 2);
  _Float16* hemith  = (_Float16*)take((size_t)S_TOT * TOKP * 2);
  _Float16* logits  = (_Float16*)take((size_t)S_TOT * VTOK * 2);

  auto grid1 = [](int total) { return dim3((total + 255) / 256); };

  // --- convert / pad (tiny, one pass each) ---
  cvt_pad_bf16_k<<<grid1(S_TOT * EMB), 256, 0, stream>>>(state_emb, S_TOT, EMB, Xbf, EMB, S_TOT * EMB);
  cvt_pad_bf16_k<<<grid1(EMB * EMB), 256, 0, stream>>>(start_w1, EMB, EMB, sW1, EMB, EMB * EMB);
  cvt_pad_bf16_k<<<grid1(EMB * EMB), 256, 0, stream>>>(start_w2, EMB, EMB, sW2, EMB, EMB * EMB);
  cvt_pad_bf16_k<<<grid1(EMB * EMB), 256, 0, stream>>>(tin_w1, EMB, EMB, iW1, EMB, EMB * EMB);
  cvt_pad_bf16_k<<<grid1(EMB * EMB), 256, 0, stream>>>(tin_w2, EMB, EMB, iW2, EMB, EMB * EMB);
  cvt_pad_bf16_k<<<grid1(EMB * EMB), 256, 0, stream>>>(tout_w1, EMB, EMB, oW1, EMB, EMB * EMB);
  cvt_pad_bf16_k<<<grid1(EMB * EMB), 256, 0, stream>>>(tout_w2, EMB, EMB, oW2, EMB, EMB * EMB);
  cvt_pad_bf16_k<<<grid1(TOKP * EMB), 256, 0, stream>>>(emit_w1, TOKD, EMB, eW1, EMB, TOKP * EMB);
  cvt_pad_bf16_k<<<grid1(TOKP * TOKP), 256, 0, stream>>>(emit_w2, TOKD, TOKD, eW2, TOKP, TOKP * TOKP);
  pad_vec_k<<<1, 256, 0, stream>>>(emit_b1, TOKD, eB1, TOKP);
  pad_vec_k<<<1, 256, 0, stream>>>(emit_b2, TOKD, eB2, TOKP);
  cvt_pad_f16_k<<<grid1(VTOK * TOKP), 256, 0, stream>>>(tok_emb, VTOK, TOKD, tokh, TOKP, VTOK * TOKP);

  // --- residual MLP blocks (bf16 WMMA) ---
  residual_kernel<EMB, EMB, EMB><<<128, 128, 0, stream>>>(Xbf, sW1, start_b1, sW2, start_b2,
                                                          start_g, start_beta, h0);
  residual_kernel<EMB, EMB, EMB><<<128, 128, 0, stream>>>(Xbf, iW1, tin_b1, iW2, tin_b2,
                                                          tin_g, tin_beta, hin);
  residual_kernel<EMB, EMB, EMB><<<128, 128, 0, stream>>>(Xbf, oW1, tout_b1, oW2, tout_b2,
                                                          tout_g, tout_beta, hout);
  residual_kernel<EMB, TOKD, TOKP><<<128, 128, 0, stream>>>(Xbf, eW1, eB1, eW2, eB2,
                                                            emit_g, emit_beta, hemit);
  cvt_pad_f16_k<<<grid1(S_TOT * TOKP), 256, 0, stream>>>(hemit, S_TOT, TOKP, hemith, TOKP, S_TOT * TOKP);

  // --- heads ---
  priors_kernel<<<1, 1024, 0, stream>>>(h0, start_wo, start_bo, priors);
  trans7_kernel<<<8, 256, 0, stream>>>(hin, hout, trans7);

  // --- emission: f16 WMMA logits (A register-blocked) -> L2-resident -> logZ -> pooled gather ---
  // 625 N-tiles, 8 tiles per wave, 8 waves per block -> 64 tiles per block -> 10 blocks in x.
  emit_logits_kernel<<<dim3(10, 128), 256, 0, stream>>>(hemith, tokh, logits);
  logz_kernel<<<S_TOT, 256, 0, stream>>>(logits, logZ);
  emis_kernel<<<(BB * TT * S_TOT + 255) / 256, 256, 0, stream>>>(logits, logZ, stories, emis);

  // --- forward recursion, single WGP, LDS-resident state ---
  forward_kernel<<<1, 1024, 0, stream>>>(emis, priors, trans7, (float*)d_out);
}
